// Net_MP_diag3_57775900066584
// MI455X (gfx1250) — compile-verified, matrix operand order checked by reading, not run on hardware
//
#include <hip/hip_runtime.h>

// CDNA5 / gfx1250, wave32. WMMA used for the dense h @ root[c] update.
typedef __attribute__((ext_vector_type(16))) _Float16 v16h;
typedef __attribute__((ext_vector_type(8)))  float    v8f;

#define TPB 256

__global__ void zero4_kernel(float4* __restrict__ p, int n4) {
  int i = blockIdx.x * blockDim.x + threadIdx.x;
  if (i < n4) p[i] = make_float4(0.f, 0.f, 0.f, 0.f);
}

__global__ void degree_kernel(const int* __restrict__ dst, float* __restrict__ cnt, int E) {
  int e = blockIdx.x * blockDim.x + threadIdx.x;
  if (e < E) atomicAdd(cnt + dst[e], 1.0f);
}

__global__ void inv_kernel(float* __restrict__ cnt, int n) {
  int i = blockIdx.x * blockDim.x + threadIdx.x;
  if (i < n) cnt[i] = 1.0f / fmaxf(cnt[i], 1.0f);
}

// h[i,:] = x[i,:] @ fc1_w.T + fc1_b    (fc1_w is [32,3] row-major)
__global__ void fc1_kernel(const float* __restrict__ x,
                           const float* __restrict__ w, const float* __restrict__ b,
                           float* __restrict__ h, int N) {
  int i = blockIdx.x * blockDim.x + threadIdx.x;
  if (i >= N) return;
  float x0 = x[3 * i], x1 = x[3 * i + 1], x2 = x[3 * i + 2];
  float* hr = h + (size_t)i * 32;
#pragma unroll
  for (int j = 0; j < 32; ++j)
    hr[j] = fmaf(w[3 * j], x0, fmaf(w[3 * j + 1], x1, fmaf(w[3 * j + 2], x2, b[j])));
}

// Per-edge: w = relu(ea @ kw1.T + kb1) @ kw2.T + kb2 ; agg[dst] += h[src] * w
// kw1:[8,3] kb1:[8] kw2:[32,8] kb2:[32]  (already offset to layer c by host code)
// Weight loads are wave-uniform -> scalar loads; h gather hits L2 (h = 6.4 MB).
__global__ void edge_scatter_kernel(const int* __restrict__ ei, const float* __restrict__ ea,
                                    const float* __restrict__ kw1, const float* __restrict__ kb1,
                                    const float* __restrict__ kw2, const float* __restrict__ kb2,
                                    const float* __restrict__ h, float* __restrict__ agg, int E) {
  int e = blockIdx.x * blockDim.x + threadIdx.x;
  if (e >= E) return;
  int s = ei[e];
  int d = ei[E + e];
  float a0 = ea[3 * e], a1 = ea[3 * e + 1], a2 = ea[3 * e + 2];
  float hid[8];
#pragma unroll
  for (int k = 0; k < 8; ++k) {
    float v = fmaf(kw1[3 * k], a0, fmaf(kw1[3 * k + 1], a1, fmaf(kw1[3 * k + 2], a2, kb1[k])));
    hid[k] = fmaxf(v, 0.f);
  }
  const float4* hs = (const float4*)(h + (size_t)s * 32);
  float4 hv[8];
#pragma unroll
  for (int q = 0; q < 8; ++q) hv[q] = hs[q];
  const float* hvf = (const float*)hv;
  float* ag = agg + (size_t)d * 32;
#pragma unroll
  for (int j = 0; j < 32; ++j) {
    float wj = kb2[j];
#pragma unroll
    for (int k = 0; k < 8; ++k) wj = fmaf(kw2[8 * j + k], hid[k], wj);
    atomicAdd(ag + j, hvf[j] * wj);  // global_atomic_add_f32, no return
  }
}

// Dense update via WMMA: h = relu(agg*inv + h @ root[c] + cbias[c]), in place.
// One wave per 16-row tile. A = h-tile (16x32 f16), B = root (32x32 f16 as two
// 32x16 halves), D = 16x16 f32 x2. Layouts per CDNA5 ISA 7.12.2:
//   A (16-bit 16x32): lane m=lane&15, hi=lane>>4; vgpr v holds K pair
//        Kv = (v<4 ? 2v : 16+2(v-4)) + (hi ? 8 : 0)
//   B mirrored with N = lane&15 across lanes, same K-pair striping.
//   C/D f32 16x16: N = lane&15, M = v + 8*hi.
__global__ __launch_bounds__(128) void update_wmma_kernel(
    float* __restrict__ h, const float* __restrict__ agg, const float* __restrict__ inv,
    const float* __restrict__ rootc, const float* __restrict__ cbiasc, int ntiles) {
  int wave = threadIdx.x >> 5;
  int lane = threadIdx.x & 31;
  int tile = blockIdx.x * (blockDim.x >> 5) + wave;
  if (tile >= ntiles) return;  // wave-uniform; active waves keep EXEC all-1s
  int m = lane & 15;
  int hi = lane >> 4;

  v16h A, B0, B1;
  const float* hrow = h + ((size_t)tile * 16 + m) * 32;
#pragma unroll
  for (int v = 0; v < 8; ++v) {
    int kb = ((v < 4) ? 2 * v : 16 + 2 * (v - 4)) + (hi ? 8 : 0);
    float2 hv = *(const float2*)(hrow + kb);
    A[2 * v]     = (_Float16)hv.x;
    A[2 * v + 1] = (_Float16)hv.y;
    B0[2 * v]     = (_Float16)rootc[kb * 32 + m];
    B0[2 * v + 1] = (_Float16)rootc[(kb + 1) * 32 + m];
    B1[2 * v]     = (_Float16)rootc[kb * 32 + m + 16];
    B1[2 * v + 1] = (_Float16)rootc[(kb + 1) * 32 + m + 16];
  }
  v8f c0 = {};
  v8f c1 = {};
  c0 = __builtin_amdgcn_wmma_f32_16x16x32_f16(false, A, false, B0, (short)0, c0, false, false);
  c1 = __builtin_amdgcn_wmma_f32_16x16x32_f16(false, A, false, B1, (short)0, c1, false, false);

  float bias0 = cbiasc[m];
  float bias1 = cbiasc[m + 16];
#pragma unroll
  for (int v = 0; v < 8; ++v) {
    size_t r = (size_t)tile * 16 + v + 8 * hi;
    float ic = inv[r];
    float o0 = fmaxf(fmaf(agg[r * 32 + m], ic, c0[v] + bias0), 0.f);
    float o1 = fmaxf(fmaf(agg[r * 32 + m + 16], ic, c1[v] + bias1), 0.f);
    h[r * 32 + m] = o0;        // all h loads completed before WMMA; in-place safe
    h[r * 32 + m + 16] = o1;
  }
}

__global__ void fc2_kernel(const float* __restrict__ h, const float* __restrict__ w,
                           const float* __restrict__ b, float* __restrict__ out, int N) {
  int i = blockIdx.x * blockDim.x + threadIdx.x;
  if (i >= N) return;
  const float4* hr = (const float4*)(h + (size_t)i * 32);
  float acc = b[0];
#pragma unroll
  for (int q = 0; q < 8; ++q) {
    float4 hv = hr[q];
    const float* wq = w + 4 * q;
    acc = fmaf(hv.x, wq[0], acc);
    acc = fmaf(hv.y, wq[1], acc);
    acc = fmaf(hv.z, wq[2], acc);
    acc = fmaf(hv.w, wq[3], acc);
  }
  out[i] = acc;
}

extern "C" void kernel_launch(void* const* d_in, const int* in_sizes, int n_in,
                              void* d_out, int out_size, void* d_ws, size_t ws_size,
                              hipStream_t stream) {
  (void)n_in; (void)out_size; (void)ws_size;
  const float* x     = (const float*)d_in[0];
  const int*   ei    = (const int*)d_in[1];
  const float* ea    = (const float*)d_in[2];
  const float* fc1_w = (const float*)d_in[3];
  const float* fc1_b = (const float*)d_in[4];
  const float* fc2_w = (const float*)d_in[5];
  const float* fc2_b = (const float*)d_in[6];
  const float* kw1   = (const float*)d_in[7];   // [3, 8, 3]
  const float* kb1   = (const float*)d_in[8];   // [3, 8]
  const float* kw2   = (const float*)d_in[9];   // [3, 32, 8]
  const float* kb2   = (const float*)d_in[10];  // [3, 32]
  const float* root  = (const float*)d_in[11];  // [3, 32, 32]
  const float* cbias = (const float*)d_in[12];  // [3, 32]

  const int N = in_sizes[0] / 3;
  const int E = in_sizes[1] / 2;
  const int ntiles = (N + 15) / 16;
  const int N16 = ntiles * 16;  // row-padded so WMMA tiles never touch unowned memory

  float* ws  = (float*)d_ws;
  float* inv = ws;                                   // [N16]   degree counts -> 1/max(c,1)
  size_t off = ((size_t)N16 + 127) & ~(size_t)127;
  float* h   = ws + off;                             // [N16, 32]
  float* agg = h + (size_t)N16 * 32;                 // [N16, 32]
  float* out = (float*)d_out;                        // [N]

  const int gN  = (N + TPB - 1) / TPB;
  const int gE  = (E + TPB - 1) / TPB;
  const int gZc = (N16 / 4 + TPB - 1) / TPB;         // N16 divisible by 4
  const int gZa = (N16 * 32 / 4 + TPB - 1) / TPB;
  const int gU  = (ntiles + 3) / 4;                  // 4 waves / block

  // in-degree -> inverse counts
  zero4_kernel<<<gZc, TPB, 0, stream>>>((float4*)inv, N16 / 4);
  degree_kernel<<<gE, TPB, 0, stream>>>(ei + E, inv, E);
  inv_kernel<<<(N16 + TPB - 1) / TPB, TPB, 0, stream>>>(inv, N16);

  // h = x @ fc1_w.T + fc1_b
  fc1_kernel<<<gN, TPB, 0, stream>>>(x, fc1_w, fc1_b, h, N);

  for (int d = 0; d < 4; ++d) {
    for (int c = 0; c < 3; ++c) {
      zero4_kernel<<<gZa, TPB, 0, stream>>>((float4*)agg, N16 * 32 / 4);
      edge_scatter_kernel<<<gE, TPB, 0, stream>>>(
          ei, ea, kw1 + c * 24, kb1 + c * 8, kw2 + c * 256, kb2 + c * 32, h, agg, E);
      update_wmma_kernel<<<gU, 128, 0, stream>>>(
          h, agg, inv, root + c * 1024, cbias + c * 32, ntiles);
    }
  }

  fc2_kernel<<<gN, TPB, 0, stream>>>(h, fc2_w, fc2_b, out, N);
}
